// GCNWithNAE_71897752535776
// MI455X (gfx1250) — compile-verified, matrix operand order checked by reading, not run on hardware
//
#include <hip/hip_runtime.h>
#include <hip/hip_bf16.h>

// ---------------------------------------------------------------------------
// GCN forward on gfx1250 (MI455X): A_norm materialized once in bf16 (128 MB,
// fits in 192 MB L2), both N x N GEMMs run on v_wmma_f32_16x16x32_bf16 with
// fp32 accumulation. The big GEMMs stage A/B k-slabs into LDS with a
// double-buffered global_load_async_to_lds_b128 pipeline (ASYNCcnt), and the
// WMMA fragments are fed from LDS via ds_load_b128 (padded, bank-conflict-free).
// ---------------------------------------------------------------------------

typedef __bf16 bf16_t;
typedef __attribute__((ext_vector_type(16))) __bf16 v16bf;
typedef __attribute__((ext_vector_type(8)))  __bf16 v8bf;
typedef __attribute__((ext_vector_type(8)))  float  v8f;
typedef __attribute__((ext_vector_type(4)))  float  v4f;

#define GCN_N    8192
#define GCN_DIN  512
#define GCN_DH   256
#define GCN_DOUT 128
#define GCN_BETA 0.8f
#define GCN_EPS  1e-8f

#define TILE_K   32
#define PAD_LD   40   // LDS row stride in elements: 80 B = 20 banks (conflict-free), 16B-aligned

// ---------------------------------------------------------------------------
// WMMA helpers
// ---------------------------------------------------------------------------

__device__ __forceinline__ v8f wmma_bf16(v16bf a, v16bf b, v8f c) {
    // (neg_a, A, neg_b, B, c_mod, C, reuse_a, reuse_b)
    return __builtin_amdgcn_wmma_f32_16x16x32_bf16(false, a, false, b,
                                                   (short)0, c, false, false);
}

// Fragment (16x32 bf16, A-layout; B identical when stored K-contiguous) from
// global memory: two 16B loads per lane.
__device__ __forceinline__ v16bf load_frag_g(const bf16_t* __restrict__ p,
                                             int ld, int base, int k, int lane) {
    const int lo = lane & 15;
    const int kh = lane >> 4;
    const bf16_t* q = p + (size_t)(base + lo) * (size_t)ld + (size_t)(k + kh * 8);
    v8bf a = *(const v8bf*)q;
    v8bf b = *(const v8bf*)(q + 16);
    return __builtin_shufflevector(a, b, 0, 1, 2, 3, 4, 5, 6, 7,
                                         8, 9, 10, 11, 12, 13, 14, 15);
}

// Same fragment from an LDS tile (row stride PAD_LD elements): two ds_load_b128.
__device__ __forceinline__ v16bf load_frag_lds(const bf16_t* p) {
    typedef __attribute__((address_space(3))) const v8bf* lds_v8bf_cp;
    lds_v8bf_cp q = (lds_v8bf_cp)p;
    v8bf a = q[0];    // k+0..7   (per half-wave)
    v8bf b = q[2];    // k+16..23 (+32 bytes)
    return __builtin_shufflevector(a, b, 0, 1, 2, 3, 4, 5, 6, 7,
                                         8, 9, 10, 11, 12, 13, 14, 15);
}

// LDS byte offset (workgroup-relative) of a __shared__ element, for async-DMA vdst.
__device__ __forceinline__ unsigned int lds_off(const bf16_t* p) {
    typedef __attribute__((address_space(3))) const bf16_t* lds_bf16_cp;
    return (unsigned int)(unsigned long long)(lds_bf16_cp)p;
}

// One 16-byte async global->LDS transfer (VGLOBAL encoding, ASYNCcnt-tracked).
__device__ __forceinline__ void async_ld16(unsigned int lds, const bf16_t* g) {
    asm volatile("global_load_async_to_lds_b128 %0, %1, off"
                 :: "v"(lds), "v"(g) : "memory");
}

__device__ __forceinline__ void wait_async0() {
    asm volatile("s_wait_asynccnt 0x0" ::: "memory");
}

// ---------------------------------------------------------------------------
// Kernel 1: per-row degree -> d^{-1/2}.  d[i] = sum_j(A+0.8S) + EPS (diag).
// ---------------------------------------------------------------------------
__global__ void __launch_bounds__(256)
rowsum_dinv(const float* __restrict__ A, const float* __restrict__ S,
            float* __restrict__ dinv, int n) {
    const int row = blockIdx.x;
    const float* ar = A + (size_t)row * n;
    const float* sr = S + (size_t)row * n;
    float sum = 0.0f;
    for (int j = threadIdx.x * 4; j < n; j += 256 * 4) {
        v4f a = *(const v4f*)(ar + j);
        v4f s = *(const v4f*)(sr + j);
        v4f t = a + GCN_BETA * s;
        sum += t[0] + t[1] + t[2] + t[3];
    }
    __shared__ float red[256];
    red[threadIdx.x] = sum;
    __syncthreads();
    for (int off = 128; off > 0; off >>= 1) {
        if (threadIdx.x < off) red[threadIdx.x] += red[threadIdx.x + off];
        __syncthreads();
    }
    if (threadIdx.x == 0) {
        float d = red[0] + GCN_EPS;
        dinv[row] = (d > 0.0f) ? rsqrtf(d) : 0.0f;
    }
}

// ---------------------------------------------------------------------------
// Kernel 2: A_norm (bf16) = dinv[i] * (A + 0.8*S + EPS*diag) * dinv[j]
// ---------------------------------------------------------------------------
__global__ void __launch_bounds__(256)
normalize_to_bf16(const float* __restrict__ A, const float* __restrict__ S,
                  const float* __restrict__ dinv, bf16_t* __restrict__ Abf,
                  int n) {
    const int row = blockIdx.y;
    const int j0  = (blockIdx.x * 256 + threadIdx.x) * 8;
    if (j0 >= n) return;
    const float di = dinv[row];
    const float* ar = A + (size_t)row * n + j0;
    const float* sr = S + (size_t)row * n + j0;
    v4f a0 = *(const v4f*)ar,          a1 = *(const v4f*)(ar + 4);
    v4f s0 = *(const v4f*)sr,          s1 = *(const v4f*)(sr + 4);
    v4f d0 = *(const v4f*)(dinv + j0), d1 = *(const v4f*)(dinv + j0 + 4);
    v8bf o;
#pragma unroll
    for (int t = 0; t < 8; t++) {
        float av = (t < 4) ? a0[t] : a1[t - 4];
        float sv = (t < 4) ? s0[t] : s1[t - 4];
        float dj = (t < 4) ? d0[t] : d1[t - 4];
        float v  = av + GCN_BETA * sv;
        if (j0 + t == row) v += GCN_EPS;
        o[t] = (bf16_t)(di * v * dj);
    }
    *(v8bf*)(Abf + (size_t)row * n + j0) = o;
}

// ---------------------------------------------------------------------------
// Kernel 3: fp32 -> bf16 elementwise (n multiple of 8)
// ---------------------------------------------------------------------------
__global__ void __launch_bounds__(256)
f32_to_bf16(const float* __restrict__ in, bf16_t* __restrict__ out, long n) {
    long i = ((long)blockIdx.x * 256 + threadIdx.x) * 8;
    if (i >= n) return;
    v4f a = *(const v4f*)(in + i);
    v4f b = *(const v4f*)(in + i + 4);
    v8bf o;
#pragma unroll
    for (int t = 0; t < 8; t++) o[t] = (bf16_t)((t < 4) ? a[t] : b[t - 4]);
    *(v8bf*)(out + i) = o;
}

// ---------------------------------------------------------------------------
// Kernel 4: fp32 RxC -> bf16 CxR transpose-convert (tiny weight matrices)
// ---------------------------------------------------------------------------
__global__ void __launch_bounds__(256)
transpose_to_bf16(const float* __restrict__ in, bf16_t* __restrict__ out,
                  int R, int C) {
    int idx = blockIdx.x * 256 + threadIdx.x;
    if (idx >= R * C) return;
    int r = idx / C, c = idx % C;
    out[(size_t)c * R + r] = (bf16_t)in[idx];
}

// ---------------------------------------------------------------------------
// Kernel 5: CT = (A @ BT^T)^T as bf16 (small GEMMs; direct global loads).
//   A  : M x K bf16 row-major,  BT : N x K bf16,  CT : N x M bf16
// Block = 8 waves (4x2) -> 128x128 tile; wave = 32x64 (2x4 WMMA tiles).
// ---------------------------------------------------------------------------
__global__ void __launch_bounds__(256)
gemm_nt_transpose_out(const bf16_t* __restrict__ A, const bf16_t* __restrict__ BT,
                      bf16_t* __restrict__ CT, int M, int N, int K) {
    const int lane = threadIdx.x & 31;
    const int wave = threadIdx.x >> 5;
    const int wr = wave >> 1, wc = wave & 1;
    const int row0 = blockIdx.x * 128 + wr * 32;
    const int col0 = blockIdx.y * 128 + wc * 64;

    v8f acc[2][4] = {};
    for (int k = 0; k < K; k += TILE_K) {
        v16bf af[2], bf[4];
#pragma unroll
        for (int i = 0; i < 2; i++) af[i] = load_frag_g(A, K, row0 + i * 16, k, lane);
#pragma unroll
        for (int j = 0; j < 4; j++) bf[j] = load_frag_g(BT, K, col0 + j * 16, k, lane);
#pragma unroll
        for (int i = 0; i < 2; i++)
#pragma unroll
            for (int j = 0; j < 4; j++)
                acc[i][j] = wmma_bf16(af[i], bf[j], acc[i][j]);
    }

    const int lo = lane & 15, kh = lane >> 4;
#pragma unroll
    for (int i = 0; i < 2; i++) {
#pragma unroll
        for (int j = 0; j < 4; j++) {
            const int n  = col0 + j * 16 + lo;
            const int m0 = row0 + i * 16 + kh * 8;
            v8bf pk;
#pragma unroll
            for (int v = 0; v < 8; v++) pk[v] = (bf16_t)acc[i][j][v];
            *(v8bf*)(CT + (size_t)n * M + m0) = pk;   // 16B aligned (m0 % 8 == 0)
        }
    }
}

// ---------------------------------------------------------------------------
// Kernel 6: C = A @ BT^T + bias[n], optional ReLU.  (the two big N x N GEMMs)
//   out_bf != nullptr -> bf16 row-major (hidden layer H), else fp32 (output).
//
// LDS-staged, double-buffered async pipeline:
//   - per k-step a 128x32 A slab and 128x32 B slab are DMA'd into LDS once
//     (global_load_async_to_lds_b128, 4 chunks/thread), removing the 2x/4x
//     redundant per-wave global loads;
//   - s_wait_asynccnt 0 + barrier, then next slab's DMA is issued while the
//     current slab feeds 8 WMMAs/wave via ds_load_b128;
//   - LDS rows padded to 80 B so half-wave b128 reads are bank-conflict-free.
// ---------------------------------------------------------------------------
__global__ void __launch_bounds__(256)
gemm_nt_bias(const bf16_t* __restrict__ A, const bf16_t* __restrict__ BT,
             const float* __restrict__ bias, bf16_t* __restrict__ out_bf,
             float* __restrict__ out_f, int M, int N, int K, int do_relu) {
    __shared__ bf16_t smA[2][128 * PAD_LD];
    __shared__ bf16_t smB[2][128 * PAD_LD];

    const int tid  = threadIdx.x;
    const int lane = tid & 31;
    const int wave = tid >> 5;
    const int wr = wave >> 1, wc = wave & 1;
    const int row0b = blockIdx.x * 128;
    const int col0b = blockIdx.y * 128;

    // DMA chunk assignment: 512 x 16B chunks per slab, 2 per thread per matrix.
    const int rA = tid >> 2;             // tile row 0..63 (and +64)
    const int q  = (tid & 3) * 8;        // element offset within the 32-wide slab
    const bf16_t* gA0 = A  + (size_t)(row0b + rA) * K + q;
    const bf16_t* gA1 = gA0 + (size_t)64 * K;
    const bf16_t* gB0 = BT + (size_t)(col0b + rA) * K + q;
    const bf16_t* gB1 = gB0 + (size_t)64 * K;

    unsigned int lA0[2], lA1[2], lB0[2], lB1[2];
#pragma unroll
    for (int b = 0; b < 2; b++) {
        lA0[b] = lds_off(&smA[b][rA * PAD_LD + q]);
        lA1[b] = lds_off(&smA[b][(rA + 64) * PAD_LD + q]);
        lB0[b] = lds_off(&smB[b][rA * PAD_LD + q]);
        lB1[b] = lds_off(&smB[b][(rA + 64) * PAD_LD + q]);
    }

    // prologue: slab 0 into buffer 0
    async_ld16(lA0[0], gA0);
    async_ld16(lA1[0], gA1);
    async_ld16(lB0[0], gB0);
    async_ld16(lB1[0], gB1);

    v8f acc[2][4] = {};
    const int lo = lane & 15, kh = lane >> 4;
    const int nk = K / TILE_K;
    for (int kt = 0; kt < nk; kt++) {
        const int cur = kt & 1;
        wait_async0();        // my slab-`cur` DMA complete
        __syncthreads();      // everyone's DMA complete & prior reads retired

        if (kt + 1 < nk) {    // overlap next slab's DMA with this slab's math
            const int nb = cur ^ 1;
            const int ko = (kt + 1) * TILE_K;
            async_ld16(lA0[nb], gA0 + ko);
            async_ld16(lA1[nb], gA1 + ko);
            async_ld16(lB0[nb], gB0 + ko);
            async_ld16(lB1[nb], gB1 + ko);
        }

        v16bf af[2], bf[4];
#pragma unroll
        for (int i = 0; i < 2; i++)
            af[i] = load_frag_lds(&smA[cur][(wr * 32 + i * 16 + lo) * PAD_LD + kh * 8]);
#pragma unroll
        for (int j = 0; j < 4; j++)
            bf[j] = load_frag_lds(&smB[cur][(wc * 64 + j * 16 + lo) * PAD_LD + kh * 8]);
#pragma unroll
        for (int i = 0; i < 2; i++)
#pragma unroll
            for (int j = 0; j < 4; j++)
                acc[i][j] = wmma_bf16(af[i], bf[j], acc[i][j]);
        // all ds_loads are consumed by the WMMAs above, so reads of this buffer
        // are retired before any wave can reach the next iteration's barrier.
    }

#pragma unroll
    for (int i = 0; i < 2; i++) {
#pragma unroll
        for (int j = 0; j < 4; j++) {
            const int n  = col0b + wc * 64 + j * 16 + lo;
            const int m0 = row0b + wr * 32 + i * 16 + kh * 8;
            const float bv = bias[n];
#pragma unroll
            for (int v = 0; v < 8; v++) {
                float val = acc[i][j][v] + bv;
                if (do_relu) val = fmaxf(val, 0.0f);
                const size_t idx = (size_t)(m0 + v) * N + n;
                if (out_bf) out_bf[idx] = (bf16_t)val;
                else        out_f[idx]  = val;
            }
        }
    }
}

// ---------------------------------------------------------------------------
// Host-side orchestration
// ---------------------------------------------------------------------------
extern "C" void kernel_launch(void* const* d_in, const int* in_sizes, int n_in,
                              void* d_out, int out_size, void* d_ws, size_t ws_size,
                              hipStream_t stream) {
    (void)in_sizes; (void)n_in; (void)out_size; (void)ws_size;
    const float* A  = (const float*)d_in[0];
    const float* S  = (const float*)d_in[1];
    const float* x  = (const float*)d_in[2];
    const float* W1 = (const float*)d_in[3];
    const float* b1 = (const float*)d_in[4];
    const float* W2 = (const float*)d_in[5];
    const float* b2 = (const float*)d_in[6];
    float* out = (float*)d_out;

    const int N = GCN_N, DIN = GCN_DIN, DH = GCN_DH, DOUT = GCN_DOUT;

    // Workspace carve-out (~146 MB total)
    char*  ws  = (char*)d_ws;
    size_t off = 0;
    auto carve = [&](size_t bytes) -> char* {
        char* p = ws + off;
        off += (bytes + 255) & ~(size_t)255;
        return p;
    };
    bf16_t* Abf  = (bf16_t*)carve((size_t)N * N    * 2);  // 128 MB, fits in L2
    bf16_t* xbf  = (bf16_t*)carve((size_t)N * DIN  * 2);  // 8 MB
    bf16_t* W1T  = (bf16_t*)carve((size_t)DH * DIN * 2);  // 256 KB
    bf16_t* W2T  = (bf16_t*)carve((size_t)DOUT * DH * 2); // 64 KB
    bf16_t* XW1T = (bf16_t*)carve((size_t)DH * N   * 2);  // 4 MB
    bf16_t* Hbf  = (bf16_t*)carve((size_t)N * DH   * 2);  // 4 MB
    bf16_t* HW2T = (bf16_t*)carve((size_t)DOUT * N * 2);  // 2 MB
    float*  dinv = (float*)carve((size_t)N * 4);          // 32 KB

    // 1) degrees -> d^{-1/2}
    rowsum_dinv<<<N, 256, 0, stream>>>(A, S, dinv, N);

    // 2) A_norm in bf16
    normalize_to_bf16<<<dim3(N / (256 * 8), N), 256, 0, stream>>>(A, S, dinv, Abf, N);

    // 3) operand conversions
    f32_to_bf16<<<(int)(((long)N * DIN / 8 + 255) / 256), 256, 0, stream>>>(x, xbf, (long)N * DIN);
    transpose_to_bf16<<<(DIN * DH + 255) / 256, 256, 0, stream>>>(W1, W1T, DIN, DH);
    transpose_to_bf16<<<(DH * DOUT + 255) / 256, 256, 0, stream>>>(W2, W2T, DH, DOUT);

    // 4) XW1^T = (x @ W1)^T          [DH x N]
    gemm_nt_transpose_out<<<dim3(N / 128, DH / 128), 256, 0, stream>>>(xbf, W1T, XW1T, N, DH, DIN);

    // 5) H = relu(A_norm @ xW1 + b1) [N x DH], bf16
    gemm_nt_bias<<<dim3(N / 128, DH / 128), 256, 0, stream>>>(Abf, XW1T, b1, Hbf, nullptr, N, DH, N, 1);

    // 6) HW2^T = (H @ W2)^T          [DOUT x N]
    gemm_nt_transpose_out<<<dim3(N / 128, DOUT / 128), 256, 0, stream>>>(Hbf, W2T, HW2T, N, DOUT, DH);

    // 7) out = A_norm @ hW2 + b2     [N x DOUT], fp32
    gemm_nt_bias<<<dim3(N / 128, DOUT / 128), 256, 0, stream>>>(Abf, HW2T, b2, nullptr, out, N, DOUT, N, 0);
}
